// GCN_RAND_LABELED_49022756716615
// MI455X (gfx1250) — compile-verified, matrix operand order checked by reading, not run on hardware
//
#include <hip/hip_runtime.h>
#include <hip/hip_bf16.h>
#include <stdint.h>
#include <stddef.h>

// GCN with random labeled sampling — CDNA5 (gfx1250, wave32) implementation.
// Dense GEMMs use v_wmma_f32_16x16x32_bf16; SpMM is edge-parallel f32 atomics
// (entire working set ~95MB fits MI455X's 192MB L2).

typedef __bf16 bf16;
typedef __attribute__((ext_vector_type(16))) __bf16 v16bf;
typedef __attribute__((ext_vector_type(8)))  __bf16 v8bf;
typedef __attribute__((ext_vector_type(8)))  float  v8f;

#define R_ 4
#define F_ 128
#define H_ 64
#define C_ 16

// ---------------- utility kernels ----------------

__global__ void k_zero(float* __restrict__ p, size_t n) {
  size_t i = (size_t)blockIdx.x * blockDim.x + threadIdx.x;
  size_t stride = (size_t)gridDim.x * blockDim.x;
  for (; i < n; i += stride) p[i] = 0.0f;
}

__global__ void k_cvt_x(const float* __restrict__ x, bf16* __restrict__ xb, size_t n) {
  size_t i = (size_t)blockIdx.x * blockDim.x + threadIdx.x;
  size_t stride = (size_t)gridDim.x * blockDim.x;
  for (; i < n; i += stride) xb[i] = (bf16)x[i];
}

// Pack W1[0:128,0:64] (row-major f32) into CDNA5 wave32 bf16 B-fragments.
// Fragment f = kt*4 + nt covers K=[kt*32,kt*32+32), N=[nt*16,nt*16+16).
// Lane l holds column n = nt*16 + (l&15); K chunking: kbase = kt*32 + (l>>4)*8,
// element j<8 -> k = kbase+j, j>=8 -> k = kbase+16+(j-8)  (i.e. kbase + (j<8?j:j+8)).
__global__ void k_pack_b1(const float* __restrict__ W1, bf16* __restrict__ B1) {
  int t = blockIdx.x * blockDim.x + threadIdx.x;          // 16 frags * 512
  if (t >= 16 * 512) return;
  int f = t >> 9, lane = (t >> 4) & 31, j = t & 15;
  int kt = f >> 2, nt = f & 3;
  int kbase = kt * 32 + (lane >> 4) * 8;
  int k = kbase + ((j < 8) ? j : (j + 8));
  int n = nt * 16 + (lane & 15);
  B1[t] = (bf16)W1[(size_t)k * H_ + n];
}

__global__ void k_pack_b2(const float* __restrict__ W2, bf16* __restrict__ B2) {
  int t = blockIdx.x * blockDim.x + threadIdx.x;          // 2 frags * 512
  if (t >= 2 * 512) return;
  int f = t >> 9, lane = (t >> 4) & 31, j = t & 15;
  int kbase = f * 32 + (lane >> 4) * 8;
  int k = kbase + ((j < 8) ? j : (j + 8));
  int n = lane & 15;
  B2[t] = (bf16)W2[(size_t)k * C_ + n];
}

// ---------------- sampling ----------------

__device__ __forceinline__ uint32_t hash3(uint32_t a, uint32_t b, uint32_t c) {
  uint32_t h = a * 0x9E3779B1u ^ b * 0x85EBCA77u ^ c * 0xC2B2AE3Du;
  h ^= h >> 16; h *= 0x7FEB352Du; h ^= h >> 15; h *= 0x846CA68Bu; h ^= h >> 16;
  return h;
}

// Gumbel-argmax categorical draw, deterministic stateless RNG keyed by (r,n,c).
__global__ void k_sample(const float* __restrict__ pred, int* __restrict__ samp, int N) {
  int idx = blockIdx.x * blockDim.x + threadIdx.x;   // idx = r*N + n
  if (idx >= R_ * N) return;
  int r = idx / N, n = idx - r * N;
  const float* p = pred + (size_t)n * C_;
  float best = -1e30f; int bi = 0;
  #pragma unroll
  for (int c = 0; c < C_; ++c) {
    uint32_t h = hash3(42u + (uint32_t)r * 0x632BE5ABu, (uint32_t)n, (uint32_t)c);
    float u = ((float)h + 0.5f) * (1.0f / 4294967296.0f);
    float g = -__logf(-__logf(u));
    float s = __logf(p[c] + 1e-12f) + g;
    if (s > best) { best = s; bi = c; }
  }
  samp[idx] = bi;
}

__global__ void k_label(const int* __restrict__ idxl, const int* __restrict__ labels,
                        int* __restrict__ samp, int N, int nl) {
  int i = blockIdx.x * blockDim.x + threadIdx.x;
  if (i >= R_ * nl) return;
  int r = i & 3, j = i >> 2;
  int node = idxl[j];
  samp[r * N + node] = labels[node];
}

// ---------------- GEMM1: XW = x @ W1[0:128,:]  via WMMA ----------------
// One wave computes a 16x64 output strip; 4 k-steps x 4 n-tiles = 16 wmma.

__global__ __launch_bounds__(32) void k_gemm1(const bf16* __restrict__ Xbf,
                                              const bf16* __restrict__ B1,
                                              float* __restrict__ XW, int N) {
  int strip = blockIdx.x;
  int lane = threadIdx.x;
  int m = lane & 15, hv = lane >> 4;
  int row = strip * 16 + m;
  v8f acc[4] = {v8f{}, v8f{}, v8f{}, v8f{}};
  #pragma unroll
  for (int kt = 0; kt < 4; ++kt) {
    int koff = kt * 32 + hv * 8;
    const bf16* ap = Xbf + (size_t)row * F_ + koff;
    v8bf lo = *(const v8bf*)ap;
    v8bf hi = *(const v8bf*)(ap + 16);
    v16bf a;
    #pragma unroll
    for (int i = 0; i < 8; ++i) { a[i] = lo[i]; a[i + 8] = hi[i]; }
    #pragma unroll
    for (int nt = 0; nt < 4; ++nt) {
      v16bf b = *(const v16bf*)(B1 + (size_t)(kt * 4 + nt) * 512 + lane * 16);
      acc[nt] = __builtin_amdgcn_wmma_f32_16x16x32_bf16(
          false, a, false, b, (short)0, acc[nt], false, false);
    }
  }
  // C/D layout: VGPR i -> M = i + 8*(lane>>4), N = lane&15
  #pragma unroll
  for (int nt = 0; nt < 4; ++nt)
    #pragma unroll
    for (int i = 0; i < 8; ++i)
      XW[(size_t)(strip * 16 + i + hv * 8) * H_ + nt * 16 + m] = acc[nt][i];
}

// ---------------- SpMM1 (fused label-row gather, atomic scatter) ----------------
// hacc[r,row,:] += val * ( XW[col,:] + W1[128 + samp[r,col], :] )
// One wave per (edge, r); 2 floats per lane.

__global__ __launch_bounds__(256) void k_spmm1(const float* __restrict__ XW,
                                               const float* __restrict__ W1,
                                               const int* __restrict__ samp,
                                               const int* __restrict__ erow,
                                               const int* __restrict__ ecol,
                                               const float* __restrict__ evalv,
                                               float* __restrict__ hacc, int N, int E) {
  int pair = blockIdx.x * 8 + (threadIdx.x >> 5);
  if (pair >= E * R_) return;
  int lane = threadIdx.x & 31;
  int e = pair >> 2, r = pair & 3;        // 4 r's of one edge share a block
  int row = erow[e], col = ecol[e];
  float v = evalv[e];
  int s = samp[r * N + col];
  const float* xw = XW + (size_t)col * H_;
  const float* wl = W1 + (size_t)(F_ + s) * H_;
  float* out = hacc + ((size_t)r * N + row) * H_;
  atomicAdd(&out[lane],      v * (xw[lane]      + wl[lane]));
  atomicAdd(&out[lane + 32], v * (xw[lane + 32] + wl[lane + 32]));
}

// ---------------- GEMM2: z[r] = relu(hacc[r]+b1) @ W2  via WMMA ----------------

__global__ __launch_bounds__(32) void k_gemm2(const float* __restrict__ hacc,
                                              const bf16* __restrict__ B2,
                                              const float* __restrict__ b1,
                                              float* __restrict__ z, int N) {
  int strips = N / 16;
  int gid = blockIdx.x;
  int r = gid / strips, strip = gid - r * strips;
  int lane = threadIdx.x, m = lane & 15, hv = lane >> 4;
  int row = strip * 16 + m;
  const float* hp = hacc + ((size_t)r * N + row) * H_;
  v8f acc = {};
  #pragma unroll
  for (int kt = 0; kt < 2; ++kt) {
    int kbase = kt * 32 + hv * 8;
    v8f h0 = *(const v8f*)(hp + kbase);
    v8f h1 = *(const v8f*)(hp + kbase + 16);
    v8f c0 = *(const v8f*)(b1 + kbase);
    v8f c1 = *(const v8f*)(b1 + kbase + 16);
    v16bf a;
    #pragma unroll
    for (int i = 0; i < 8; ++i) {
      float t0 = h0[i] + c0[i]; t0 = t0 > 0.0f ? t0 : 0.0f;
      float t1 = h1[i] + c1[i]; t1 = t1 > 0.0f ? t1 : 0.0f;
      a[i] = (bf16)t0; a[i + 8] = (bf16)t1;
    }
    v16bf b = *(const v16bf*)(B2 + (size_t)kt * 512 + lane * 16);
    acc = __builtin_amdgcn_wmma_f32_16x16x32_bf16(
        false, a, false, b, (short)0, acc, false, false);
  }
  #pragma unroll
  for (int i = 0; i < 8; ++i)
    z[((size_t)r * N + strip * 16 + i + hv * 8) * C_ + m] = acc[i];
}

// ---------------- SpMM2 (mean over R folded into one accumulator) ----------------

__global__ __launch_bounds__(256) void k_spmm2(const float* __restrict__ z,
                                               const int* __restrict__ erow,
                                               const int* __restrict__ ecol,
                                               const float* __restrict__ evalv,
                                               float* __restrict__ acc2, int N, int E) {
  long idx = (long)blockIdx.x * blockDim.x + threadIdx.x;
  if (idx >= (long)E * R_ * C_) return;
  int c = (int)(idx & 15);
  long t = idx >> 4;
  int r = (int)(t & 3);
  int e = (int)(t >> 2);
  int row = erow[e], col = ecol[e];
  float v = evalv[e];
  atomicAdd(&acc2[(size_t)row * C_ + c], v * z[((size_t)r * N + col) * C_ + c]);
}

// ---------------- final: logits = acc2/R + b2, log_softmax ----------------

__global__ void k_final(const float* __restrict__ acc2, const float* __restrict__ b2,
                        float* __restrict__ out, int N) {
  int n = blockIdx.x * blockDim.x + threadIdx.x;
  if (n >= N) return;
  float v[C_]; float mx = -1e30f;
  #pragma unroll
  for (int c = 0; c < C_; ++c) {
    v[c] = acc2[(size_t)n * C_ + c] * (1.0f / R_) + b2[c];
    mx = v[c] > mx ? v[c] : mx;
  }
  float s = 0.0f;
  #pragma unroll
  for (int c = 0; c < C_; ++c) s += __expf(v[c] - mx);
  float lse = __logf(s);
  #pragma unroll
  for (int c = 0; c < C_; ++c) out[(size_t)n * C_ + c] = v[c] - mx - lse;
}

// ---------------- host launcher ----------------

static inline size_t align256(size_t x) { return (x + 255u) & ~(size_t)255u; }

extern "C" void kernel_launch(void* const* d_in, const int* in_sizes, int n_in,
                              void* d_out, int out_size, void* d_ws, size_t ws_size,
                              hipStream_t stream) {
  const float* x      = (const float*)d_in[0];
  const float* pred   = (const float*)d_in[1];
  const int*   erow   = (const int*)d_in[2];
  const int*   ecol   = (const int*)d_in[3];
  const float* evalv  = (const float*)d_in[4];
  const int*   labels = (const int*)d_in[5];
  const int*   idxl   = (const int*)d_in[6];
  const float* W1     = (const float*)d_in[7];
  const float* b1     = (const float*)d_in[8];
  const float* W2     = (const float*)d_in[9];
  const float* b2     = (const float*)d_in[10];
  float* out = (float*)d_out;

  const int N  = in_sizes[5];
  const int E  = in_sizes[2];
  const int nl = in_sizes[6];

  // workspace carve-up
  char* base = (char*)d_ws;
  size_t off = 0;
  bf16*  Xbf  = (bf16*) (base + off); off = align256(off + (size_t)N * F_ * sizeof(bf16));
  float* XW   = (float*)(base + off); off = align256(off + (size_t)N * H_ * sizeof(float));
  float* hacc = (float*)(base + off); off = align256(off + (size_t)R_ * N * H_ * sizeof(float));
  float* acc2 = (float*)(base + off); off = align256(off + (size_t)N * C_ * sizeof(float));
  float* z    = (float*)(base + off); off = align256(off + (size_t)R_ * N * C_ * sizeof(float));
  int*   samp = (int*)  (base + off); off = align256(off + (size_t)R_ * N * sizeof(int));
  bf16*  B1   = (bf16*) (base + off); off = align256(off + (size_t)16 * 512 * sizeof(bf16));
  bf16*  B2   = (bf16*) (base + off); off = align256(off + (size_t)2 * 512 * sizeof(bf16));

  // 1. zero accumulators (must happen every call)
  {
    size_t nh = (size_t)R_ * N * H_;
    k_zero<<<2048, 256, 0, stream>>>(hacc, nh);
    k_zero<<<512, 256, 0, stream>>>(acc2, (size_t)N * C_);
  }
  // 2. convert x to bf16; pack W1/W2 into WMMA B-fragment layout
  k_cvt_x<<<4096, 256, 0, stream>>>(x, Xbf, (size_t)N * F_);
  k_pack_b1<<<(16 * 512 + 255) / 256, 256, 0, stream>>>(W1, B1);
  k_pack_b2<<<(2 * 512 + 255) / 256, 256, 0, stream>>>(W2, B2);
  // 3. sample labels, overwrite labeled nodes
  k_sample<<<(R_ * N + 255) / 256, 256, 0, stream>>>(pred, samp, N);
  k_label<<<(R_ * nl + 255) / 256, 256, 0, stream>>>(idxl, labels, samp, N, nl);
  // 4. ensemble-invariant dense GEMM: XW = x @ W1[:128,:]   (WMMA)
  k_gemm1<<<N / 16, 32, 0, stream>>>(Xbf, B1, XW, N);
  // 5. gc1 SpMM with fused one-hot W1-row gather + atomic scatter
  k_spmm1<<<(E * R_ + 7) / 8, 256, 0, stream>>>(XW, W1, samp, erow, ecol, evalv, hacc, N, E);
  // 6. gc2 dense GEMM with fused bias+relu on load   (WMMA)
  k_gemm2<<<R_ * (N / 16), 32, 0, stream>>>(hacc, B2, b1, z, N);
  // 7. gc2 SpMM, mean over R folded into one accumulator
  {
    long tot = (long)E * R_ * C_;
    k_spmm2<<<(int)((tot + 255) / 256), 256, 0, stream>>>(z, erow, ecol, evalv, acc2, N, E);
  }
  // 8. bias + log_softmax
  k_final<<<(N + 255) / 256, 256, 0, stream>>>(acc2, b2, out, N);
}
